// SNN_VGG_16157666968232
// MI455X (gfx1250) — compile-verified, matrix operand order checked by reading, not run on hardware
//
#include <hip/hip_runtime.h>

typedef __attribute__((ext_vector_type(16))) _Float16     v16h;
typedef __attribute__((ext_vector_type(8)))  float        v8f;
typedef __attribute__((ext_vector_type(4)))  unsigned int v4u;
typedef __attribute__((ext_vector_type(8)))  int          v8i;
typedef __attribute__((ext_vector_type(4)))  int          v4i;

#define LEAK 0.99f
#define THR  1.0f
#define EPS  1e-4f

// ---------------------------------------------------------------------------
// TDM: async 2D tile load (global -> LDS). Descriptor per CDNA5 ISA 8.3/8.4.
// ---------------------------------------------------------------------------
__device__ __forceinline__ void tdm_load_a(unsigned ldsOff, unsigned long long ga,
                                           int rowLenH, int rows, int rowStrideH) {
    v4u g0; v8i g1;
    g0[0] = 1u;                                   // count=1, user mode
    g0[1] = ldsOff;                               // lds_addr (bytes)
    g0[2] = (unsigned)ga;                         // global_addr[31:0]
    g0[3] = (unsigned)((ga >> 32) & 0x1FFFFFFu)   // global_addr[56:32]
          | (2u << 30);                           // type = 2 ("image")
    g1[0] = (int)(1u << 16);                      // data_size = 1 -> 2 bytes
    g1[1] = 0;                                    // tensor_dim0 low16 (dim0=0x40000000)
    g1[2] = (int)(0x4000u | ((unsigned)rows << 16)); // dim0 hi16 ; tensor_dim1 = rows
    g1[3] = (int)((unsigned)rowLenH << 16);       // tile_dim0
    g1[4] = rows;                                 // tile_dim1 = rows, tile_dim2 = 0
    g1[5] = rowStrideH;                           // tensor_dim0_stride[31:0]
    g1[6] = 0;
    g1[7] = 0;
    v4i z4 = {};
#if defined(__clang_major__) && (__clang_major__ >= 23)
    v8i z8 = {};
    __builtin_amdgcn_tensor_load_to_lds(g0, g1, z4, z4, z8, 0);
#else
    __builtin_amdgcn_tensor_load_to_lds(g0, g1, z4, z4, 0);
#endif
}

// ---------------------------------------------------------------------------
// zero fill
// ---------------------------------------------------------------------------
__global__ void fill_zero_f32(float* __restrict__ p, int n) {
    int g = blockIdx.x * 256 + threadIdx.x;
    if (g < n) p[g] = 0.0f;
}

// ---------------------------------------------------------------------------
// Pack conv weights (fp32 OIHW) into WMMA A-fragment order (fp16).
// 16-bit A 16x32 layout (ISA 7.12.2): lane l row M=l&15;
//   K = (v>=4?16:0) + ((l>=16)?8:0) + (v&3)*2 + j
// Flat: [mtile][ktile][lane][16 halves]. K order: kk=(ky*3+kx)*Cin+ci (3x3).
// ---------------------------------------------------------------------------
__global__ void pack_weights(const float* __restrict__ W, _Float16* __restrict__ apack,
                             int Cout, int Cin, int K, int KT, int ksize, int total) {
    int gid = blockIdx.x * 256 + threadIdx.x;
    if (gid >= total) return;
    int h      = gid & 15;
    int lane   = (gid >> 4) & 31;
    int tileId = gid >> 9;
    int mt = tileId / KT;
    int kt = tileId - mt * KT;
    int v = h >> 1, j = h & 1;
    int kk = kt * 32 + ((v >= 4) ? 16 : 0) + ((lane >> 4) << 3) + ((v & 3) << 1) + j;
    int m  = mt * 16 + (lane & 15);
    _Float16 val = (_Float16)0.0f;
    if (m < Cout && kk < K) {
        if (ksize == 3) {
            int k9 = kk / Cin;
            int ci = kk - k9 * Cin;
            int ky = k9 / 3, kx = k9 - ky * 3;
            val = (_Float16)W[(((size_t)m * Cin + ci) * 3 + ky) * 3 + kx];
        } else {
            val = (_Float16)W[(size_t)m * Cin + kk];
        }
    }
    apack[gid] = val;
}

// ---------------------------------------------------------------------------
// Layer 0: Poisson spike generator + 3->64 conv (3x3 pad 1) + BN + LIF.
// ---------------------------------------------------------------------------
__global__ __launch_bounds__(256)
void layer0_lif(const float* __restrict__ x, const float* __restrict__ u_t,
                const float* __restrict__ W0, const float* __restrict__ bn_t,
                float* __restrict__ mem, _Float16* __restrict__ act_out) {
    int gid = blockIdx.x * 256 + threadIdx.x;        // n<<20 | co<<14 | y<<7 | x
    int xc = gid & 127;
    int yc = (gid >> 7) & 127;
    int co = (gid >> 14) & 63;
    int n  = gid >> 20;
    if (n >= 4) return;
    float acc = 0.0f;
    for (int ci = 0; ci < 3; ++ci) {
        const float* xin = x   + ((size_t)n * 3 + ci) * 16384;
        const float* uin = u_t + ((size_t)n * 3 + ci) * 16384;
        const float* w   = W0  + (co * 3 + ci) * 9;
        #pragma unroll
        for (int ky = 0; ky < 3; ++ky) {
            int yy = yc + ky - 1;
            if (yy < 0 || yy >= 128) continue;
            #pragma unroll
            for (int kx = 0; kx < 3; ++kx) {
                int xx = xc + kx - 1;
                if (xx < 0 || xx >= 128) continue;
                float xv = xin[yy * 128 + xx];
                float uv = uin[yy * 128 + xx];
                float s = (uv <= fabsf(xv))
                          ? ((xv > 0.0f) ? 1.0f : ((xv < 0.0f) ? -1.0f : 0.0f))
                          : 0.0f;
                acc += w[ky * 3 + kx] * s;
            }
        }
    }
    float g = bn_t[co], mu = bn_t[64 + co], vr = bn_t[128 + co];
    float o  = (acc - mu) * (g * rsqrtf(vr + EPS));
    float mv = LEAK * mem[gid] + o;
    float s  = (mv > THR) ? 1.0f : 0.0f;
    mem[gid] = mv - THR * s;
    act_out[gid] = (_Float16)s;
}

// ---------------------------------------------------------------------------
// AvgPool2d(3, stride 2, pad 1), count_include_pad (divide by 9). fp16.
// ---------------------------------------------------------------------------
__global__ __launch_bounds__(256)
void avgpool3x3s2(const _Float16* __restrict__ in, _Float16* __restrict__ out,
                  int C, int Win, int total) {
    int gid = blockIdx.x * 256 + threadIdx.x;
    if (gid >= total) return;
    int Wo = Win >> 1;
    int xo = gid % Wo;
    int t  = gid / Wo;
    int yo = t % Wo;  t /= Wo;
    int c  = t % C;
    int n  = t / C;
    const _Float16* src = in + (((size_t)n * C + c) * Win) * Win;
    float s = 0.0f;
    #pragma unroll
    for (int ky = 0; ky < 3; ++ky) {
        int yi = yo * 2 + ky - 1;
        if (yi < 0 || yi >= Win) continue;
        #pragma unroll
        for (int kx = 0; kx < 3; ++kx) {
            int xi = xo * 2 + kx - 1;
            if (xi < 0 || xi >= Win) continue;
            s += (float)src[yi * Win + xi];
        }
    }
    out[gid] = (_Float16)(s * (1.0f / 9.0f));
}

// ---------------------------------------------------------------------------
// Implicit-GEMM conv via WMMA f32_16x16x32_f16, fused BN + LIF.
//   M = COUT, N = pixels, K = CIN*KSIZE^2.  CIN % 32 == 0.
// Fully shape-specialized: WDIM power of two -> all index math is shift/mask.
// Workgroup = 8 waves as 4(M) x 2(N): 64 Cout x 32 contiguous-x pixels.
// Per tap-group (GCH K-chunks sharing one conv tap):
//   * wave 0 issues a TDM 2D-tile load of the 4 M-tiles' A-fragments -> LDS
//   * all threads stage the activation tile into LDS in B-fragment-major
//     order (B loads become contiguous 32B ds reads)
//   * s_wait_tensorcnt + barrier, then GCH unrolled WMMAs per wave.
// MODE 0: LIF epilogue.  MODE 1: out_acc += conv * outScale.
// ---------------------------------------------------------------------------
template <int CIN, int COUT, int WDIM, int DIL, int KSIZE, int MODE>
__global__ __launch_bounds__(256)
void conv_wmma_lif(const _Float16* __restrict__ act_in,
                   const _Float16* __restrict__ apack,
                   const float* __restrict__ bn_t,
                   float* __restrict__ mem,
                   _Float16* __restrict__ act_out,
                   float* __restrict__ out_acc,
                   float outScale) {
    constexpr int KTOT    = (CIN * KSIZE * KSIZE) / 32;   // total K chunks
    constexpr int CHUNKS  = CIN / 32;                     // chunks per conv tap
    constexpr int GCH     = (CHUNKS < 8) ? CHUNKS : 8;    // chunks per group
    constexpr int NGROUPS = KTOT / GCH;
    constexpr int TPIMG   = (WDIM * WDIM) / 32;           // pixel tiles per image

    __shared__ _Float16 bTile[GCH * 1024];                // GCH x 2 colgrp x 512
    __shared__ _Float16 aLds [GCH * 4 * 512];             // 4 mtiles x GCH x 512

    const int tid  = threadIdx.x;
    const int lane = tid & 31;
    const int wave = tid >> 5;
    const int mw   = wave >> 1;      // 0..3  (M direction)
    const int nw   = wave & 1;       // 0..1  (N direction)

    // pixel tile: 32 contiguous x in one row (n, y) -- all shifts
    const int pt  = blockIdx.x;
    const int n   = pt / TPIMG;
    const int rem = pt & (TPIMG - 1);
    const int y   = rem / (WDIM >> 5);
    const int x0  = (rem & ((WDIM >> 5) - 1)) << 5;

    const int mtile = blockIdx.y * 4 + mw;

    v8f acc = {};

    // B staging: thread handles k = tid>>3 within each chunk, 4 x positions
    const int s_k = tid >> 3;
    const int s_x = (tid & 7) << 2;
    const int sg  = s_x >> 4;                              // column group 0/1
    const int kb256 = ((s_k >> 3) & 1) << 8;               // lane-high * 16 halves
    const int vfrag = ((s_k >> 4) << 2) | ((s_k >> 1) & 3);
    const int h_st  = (vfrag << 1) | (s_k & 1);            // fragment element idx

    const unsigned aLdsOff = (unsigned)(unsigned long long)(uintptr_t)&aLds[0];
    const unsigned long long aBaseRow0 =
        (unsigned long long)(uintptr_t)(apack + (size_t)(blockIdx.y * 4) * KTOT * 512);

    for (int g = 0; g < NGROUPS; ++g) {
        const int kt0 = g * GCH;
        int ci0, yy, xb;
        if (KSIZE == 3) {
            int tap = (kt0 * 32) / CIN;                    // constexpr shift
            ci0 = kt0 * 32 - tap * CIN;
            int ky = tap / 3, kx = tap - ky * 3;
            yy = y  + DIL * (ky - 1);
            xb = x0 + DIL * (kx - 1);
        } else {
            ci0 = kt0 * 32; yy = y; xb = x0;
        }

        __syncthreads();   // previous group fully consumed

        // async DMA: 4 rows (M tiles) of GCH*512 halves, stride KTOT*512
        if (wave == 0) {
            tdm_load_a(aLdsOff,
                       aBaseRow0 + (unsigned long long)kt0 * 1024ull,
                       GCH * 512, 4, KTOT * 512);
        }

        // stage activations in B-fragment-major order (overlaps with TDM)
        #pragma unroll
        for (int cc = 0; cc < GCH; ++cc) {
            const int ci = ci0 + cc * 32 + s_k;
            _Float16* dst = &bTile[(cc * 2 + sg) * 512 + kb256 + h_st];
            if (yy >= 0 && yy < WDIM) {
                const _Float16* src =
                    act_in + (((size_t)n * CIN + ci) * WDIM + yy) * WDIM;
                #pragma unroll
                for (int q = 0; q < 4; ++q) {
                    int xx = xb + s_x + q;
                    int col = (s_x + q) & 15;
                    dst[col * 16] =
                        (xx >= 0 && xx < WDIM) ? src[xx] : (_Float16)0.0f;
                }
            } else {
                #pragma unroll
                for (int q = 0; q < 4; ++q) {
                    int col = (s_x + q) & 15;
                    dst[col * 16] = (_Float16)0.0f;
                }
            }
        }

        if (wave == 0) __builtin_amdgcn_s_wait_tensorcnt(0);
        __syncthreads();   // A (TDM) + B (ds stores) visible to all waves

        #pragma unroll
        for (int cc = 0; cc < GCH; ++cc) {
            v16h afrag = *(const v16h*)&aLds [((mw * GCH + cc) * 32 + lane) * 16];
            v16h bfrag = *(const v16h*)&bTile[((cc * 2  + nw) * 32 + lane) * 16];
            acc = __builtin_amdgcn_wmma_f32_16x16x32_f16(
                false, afrag, false, bfrag, (short)0, acc, false, false);
        }
    }

    // epilogue: C/D layout -> lane col = lane&15, rows v + 8*(lane>=16)
    const int col   = (lane & 15) + (nw << 4);
    const int px    = x0 + col;
    const int mhalf = (lane >> 4) << 3;

    if (MODE == 0) {
        #pragma unroll
        for (int v = 0; v < 8; ++v) {
            int m = mtile * 16 + mhalf + v;
            float gg = bn_t[m];
            float mu = bn_t[COUT + m];
            float vr = bn_t[2 * COUT + m];
            float o  = (acc[v] - mu) * (gg * rsqrtf(vr + EPS));
            size_t idx = (((size_t)n * COUT + m) * WDIM + y) * WDIM + px;
            float mv = LEAK * mem[idx] + o;
            float s  = (mv > THR) ? 1.0f : 0.0f;
            mem[idx] = mv - THR * s;
            act_out[idx] = (_Float16)s;
        }
    } else {
        #pragma unroll
        for (int v = 0; v < 8; ++v) {
            int m = mtile * 16 + mhalf + v;
            if (m < COUT) {
                size_t idx = (((size_t)n * COUT + m) * WDIM + y) * WDIM + px;
                out_acc[idx] += acc[v] * outScale;
            }
        }
    }
}

// ---------------------------------------------------------------------------
// Host orchestration
// ---------------------------------------------------------------------------
extern "C" void kernel_launch(void* const* d_in, const int* in_sizes, int n_in,
                              void* d_out, int out_size, void* d_ws, size_t ws_size,
                              hipStream_t stream) {
    (void)in_sizes; (void)n_in; (void)out_size; (void)ws_size;

    const float* x  = (const float*)d_in[0];
    const float* u  = (const float*)d_in[1];
    const float* Wt[9];
    for (int i = 0; i < 9; ++i) Wt[i] = (const float*)d_in[2 + i];
    const float* bn[8];
    for (int i = 0; i < 8; ++i) bn[i] = (const float*)d_in[11 + i];
    float* out = (float*)d_out;

    // ---- workspace layout ----
    static const size_t MSZ[8] = {4194304, 4194304, 2097152, 2097152,
                                  1048576, 1048576, 1048576, 4194304}; // floats
    float* memp[8];
    size_t off = 0;
    for (int i = 0; i < 8; ++i) { memp[i] = (float*)d_ws + off; off += MSZ[i]; }
    const size_t MEM_TOTAL = off;                          // 19,922,944 floats
    _Float16* actA = (_Float16*)((char*)d_ws + MEM_TOTAL * 4);
    _Float16* actB = actA + 4194304;
    _Float16* apk  = actB + 4194304;

    // layers 1..7 + classifier: {Cin, Mround, K, KT, ksize}
    struct L { int Cin, Mr, K, KT, ks; };
    static const L LL[8] = {
        { 64,   64,  576, 18, 3}, { 64,  128,  576, 18, 3},
        {128,  128, 1152, 36, 3}, {128,  256, 1152, 36, 3},
        {256,  256, 2304, 72, 3}, {256,  256, 2304, 72, 3},
        {256, 1024, 2304, 72, 3}, {1024,  64, 1024, 32, 1},
    };
    _Float16* ap[8];
    {
        size_t o2 = 0;
        for (int i = 0; i < 8; ++i) { ap[i] = apk + o2; o2 += (size_t)LL[i].Mr * LL[i].K; }
    }

    // ---- zero membrane state + output accumulator ----
    {
        int nz = (int)MEM_TOTAL;
        fill_zero_f32<<<(nz + 255) / 256, 256, 0, stream>>>((float*)d_ws, nz);
        fill_zero_f32<<<(86016 + 255) / 256, 256, 0, stream>>>(out, 86016);
    }

    // ---- pack weights into WMMA fragment order (fp16) ----
    static const int CoutReal[8] = {64, 128, 128, 256, 256, 256, 1024, 21};
    for (int i = 0; i < 8; ++i) {
        int total = LL[i].Mr * LL[i].K;
        pack_weights<<<(total + 255) / 256, 256, 0, stream>>>(
            Wt[i + 1], ap[i], CoutReal[i], LL[i].Cin, LL[i].K, LL[i].KT,
            LL[i].ks, total);
    }

    // ---- timestep loop ----
    const size_t uStride = (size_t)4 * 3 * 128 * 128;
    for (int t = 0; t < 8; ++t) {
        layer0_lif<<<16384, 256, 0, stream>>>(
            x, u + t * uStride, Wt[0], bn[0] + t * 192, memp[0], actA);

        conv_wmma_lif<64, 64, 128, 1, 3, 0><<<dim3(2048, 1), 256, 0, stream>>>(
            actA, ap[0], bn[1] + t * 192, memp[1], actB, nullptr, 0.f);
        avgpool3x3s2<<<4096, 256, 0, stream>>>(actB, actA, 64, 128, 1048576);

        conv_wmma_lif<64, 128, 64, 1, 3, 0><<<dim3(512, 2), 256, 0, stream>>>(
            actA, ap[1], bn[2] + t * 384, memp[2], actB, nullptr, 0.f);
        conv_wmma_lif<128, 128, 64, 1, 3, 0><<<dim3(512, 2), 256, 0, stream>>>(
            actB, ap[2], bn[3] + t * 384, memp[3], actA, nullptr, 0.f);
        avgpool3x3s2<<<2048, 256, 0, stream>>>(actA, actB, 128, 64, 524288);

        conv_wmma_lif<128, 256, 32, 1, 3, 0><<<dim3(128, 4), 256, 0, stream>>>(
            actB, ap[3], bn[4] + t * 768, memp[4], actA, nullptr, 0.f);
        conv_wmma_lif<256, 256, 32, 2, 3, 0><<<dim3(128, 4), 256, 0, stream>>>(
            actA, ap[4], bn[5] + t * 768, memp[5], actB, nullptr, 0.f);
        conv_wmma_lif<256, 256, 32, 2, 3, 0><<<dim3(128, 4), 256, 0, stream>>>(
            actB, ap[5], bn[6] + t * 768, memp[6], actA, nullptr, 0.f);
        conv_wmma_lif<256, 1024, 32, 12, 3, 0><<<dim3(128, 16), 256, 0, stream>>>(
            actA, ap[6], bn[7] + t * 3072, memp[7], actB, nullptr, 0.f);

        // classifier: 1x1 1024->21, out += conv / T
        conv_wmma_lif<1024, 21, 32, 0, 1, 1><<<dim3(128, 1), 256, 0, stream>>>(
            actB, ap[7], nullptr, nullptr, nullptr, out, 0.125f);
    }
}
